// GemmaAttention_23699629539997
// MI455X (gfx1250) — matrix operand order repaired and checked by reading, not compiled, not run on hardware
//
#include <hip/hip_runtime.h>
#include <hip/hip_bf16.h>

// ---------------------------------------------------------------------------
// Gemma GQA attention for MI455X (gfx1250), bf16 WMMA pipeline, f32 accumulate.
// B=2, S=2048, HID=2048, NH=8, NKV=1, HD=256.
// gfx1250 features: v_wmma_f32_16x16x32_bf16, async-to-LDS double-buffered
// A-panel staging (ASYNCcnt), WGP-scope prefetch.
// ---------------------------------------------------------------------------

#define Bc   2
#define Sc   2048
#define HIDc 2048
#define NHc  8
#define HDc  256
#define Mrows (Bc * Sc)          // 4096

typedef __attribute__((ext_vector_type(16))) __bf16 bf16x16;
typedef __attribute__((ext_vector_type(8)))  __bf16 bf16x8;
typedef __attribute__((ext_vector_type(8)))  float  f32x8;

union AFrag { bf16x16 v; bf16x8 h[2]; };

static __device__ __forceinline__ f32x8 wmma_bf16(bf16x16 a, bf16x16 b, f32x8 c) {
    return __builtin_amdgcn_wmma_f32_16x16x32_bf16(false, a, false, b, (short)0, c,
                                                   false, false);
}

// gfx1250 async copy: 16B/lane global -> LDS, tracked by ASYNCcnt.
static __device__ __forceinline__ void async_load_b128(unsigned lds_off,
                                                       const void* gptr) {
    unsigned long long ga = (unsigned long long)(uintptr_t)gptr;
    asm volatile("global_load_async_to_lds_b128 %0, %1, off"
                 :: "v"(lds_off), "v"(ga) : "memory");
}
static __device__ __forceinline__ void wait_asynccnt0() {
    asm volatile("s_wait_asynccnt 0x0" ::: "memory");
}

// ---------------------------------------------------------------------------
// 1) f32 -> bf16 straight convert
// ---------------------------------------------------------------------------
__global__ void k_conv_bf16(const float* __restrict__ in,
                            __hip_bfloat16* __restrict__ out, int n) {
    int i = blockIdx.x * blockDim.x + threadIdx.x;
    if (i < n) out[i] = __float2bfloat16(in[i]);
}

// 2) weight transpose+convert: in f32 [K,N] -> out bf16 [N,K]
__global__ void k_wtrans(const float* __restrict__ in,
                         __hip_bfloat16* __restrict__ out, int K, int N) {
    int i = blockIdx.x * blockDim.x + threadIdx.x;
    if (i >= K * N) return;
    int k = i / N, n = i - k * N;
    out[(size_t)n * K + k] = __float2bfloat16(in[i]);
}

// ---------------------------------------------------------------------------
// 3a) Direct GEMM (used for the small-N K/V projections)
//     C[M,N] = A[M,K] * B[K,N]; A bf16 [M,K], Bt bf16 [N,K], C f32.
// ---------------------------------------------------------------------------
__global__ __launch_bounds__(256) void k_gemm_bf16(
        const __hip_bfloat16* __restrict__ A,
        const __hip_bfloat16* __restrict__ Bt,
        float* __restrict__ C, int M, int N, int K) {
    const int tilesN = N >> 6;
    int wave = (blockIdx.x * blockDim.x + threadIdx.x) >> 5;
    int lane = threadIdx.x & 31;
    int tm = wave / tilesN, tn = wave - tm * tilesN;
    if (tm * 64 >= M) return;
    int m0 = tm * 64, n0 = tn * 64;
    int half = lane >> 4, l16 = lane & 15;

    const __bf16* Ab = (const __bf16*)A;
    const __bf16* Bb = (const __bf16*)Bt;

    f32x8 acc[16];
#pragma unroll
    for (int i = 0; i < 16; ++i) acc[i] = (f32x8){0,0,0,0,0,0,0,0};

    for (int kk = 0; kk < K; kk += 32) {
        AFrag a[4];
#pragma unroll
        for (int i = 0; i < 4; ++i) {
            const __bf16* ap = Ab + (size_t)(m0 + i * 16 + l16) * K + kk + half * 16;
            a[i].h[0] = *(const bf16x8*)(ap);
            a[i].h[1] = *(const bf16x8*)(ap + 8);
        }
        bf16x16 b[4];
#pragma unroll
        for (int j = 0; j < 4; ++j) {
            const __bf16* bp = Bb + (size_t)(n0 + j * 16 + l16) * K + kk + half * 16;
            b[j] = *(const bf16x16*)(bp);
            __builtin_prefetch(bp + 32, 0, 3);    // WGP-scope prefetch, next K
        }
#pragma unroll
        for (int i = 0; i < 4; ++i)
#pragma unroll
            for (int j = 0; j < 4; ++j)
                acc[i * 4 + j] = wmma_bf16(a[i].v, b[j], acc[i * 4 + j]);
    }

#pragma unroll
    for (int i = 0; i < 4; ++i)
#pragma unroll
        for (int j = 0; j < 4; ++j) {
            f32x8 c = acc[i * 4 + j];
            int col = n0 + j * 16 + l16;
#pragma unroll
            for (int r = 0; r < 8; ++r) {
                int row = m0 + i * 16 + half * 8 + r;
                C[(size_t)row * N + col] = c[r];
            }
        }
}

// ---------------------------------------------------------------------------
// 3b) LDS-staged GEMM for big-N projections (Q-proj, O-proj).
//     Block = 8 waves = 64(M) x 512(N) tile. All waves share one 64xK A panel:
//     stage it through LDS with async-to-LDS, double buffered (2 x 4KB).
//     A-side global traffic drops 8x; async fetch overlaps WMMA compute.
// ---------------------------------------------------------------------------
__global__ __launch_bounds__(256) void k_gemm_lds(
        const __hip_bfloat16* __restrict__ A,
        const __hip_bfloat16* __restrict__ Bt,
        float* __restrict__ C, int M, int N, int K) {
    __shared__ __hip_bfloat16 Apan[2][64 * 32];   // [buf][row*32 + k]

    const int nb = N >> 9;                        // blocks along N (512 wide)
    int blk = blockIdx.x;
    int tm = blk / nb;
    int m0 = tm * 64;
    int n0 = (blk - tm * nb) * 512 + (int)(threadIdx.x >> 5) * 64;
    int lane = threadIdx.x & 31;
    int half = lane >> 4, l16 = lane & 15;

    // async staging assignment: 256 threads x 16B = 4KB = one 64x32 bf16 panel
    int srow = (int)threadIdx.x >> 2;             // 0..63
    int schunk = ((int)threadIdx.x & 3) * 8;      // 0,8,16,24 (bf16 elements)
    unsigned lds0 = (unsigned)(uintptr_t)&Apan[0][srow * 32 + schunk];
    unsigned lds1 = (unsigned)(uintptr_t)&Apan[1][srow * 32 + schunk];
    const __bf16* Ab = (const __bf16*)A;
    const __bf16* Bb = (const __bf16*)Bt;
    const __bf16* agp = Ab + (size_t)(m0 + srow) * K + schunk;

    f32x8 acc[16];
#pragma unroll
    for (int i = 0; i < 16; ++i) acc[i] = (f32x8){0,0,0,0,0,0,0,0};

    const int nk = K >> 5;
    async_load_b128(lds0, agp);                   // prologue: buffer 0, kk=0

    for (int it = 0; it < nk; ++it) {
        int kk = it << 5;
        int c = it & 1;
        wait_asynccnt0();                         // my loads for buf c landed
        __syncthreads();                          // everyone's landed; prior
                                                  // reads of buf c^1 retired
        if (it + 1 < nk)                          // overlap next panel fetch
            async_load_b128(c ? lds0 : lds1, agp + (kk + 32));

        AFrag a[4];
#pragma unroll
        for (int i = 0; i < 4; ++i) {             // A fragments from LDS
            const __bf16* ap =
                (const __bf16*)&Apan[c][(i * 16 + l16) * 32 + half * 16];
            a[i].h[0] = *(const bf16x8*)(ap);
            a[i].h[1] = *(const bf16x8*)(ap + 8);
        }
        bf16x16 b[4];
#pragma unroll
        for (int j = 0; j < 4; ++j) {
            const __bf16* bp = Bb + (size_t)(n0 + j * 16 + l16) * K + kk + half * 16;
            b[j] = *(const bf16x16*)(bp);
            __builtin_prefetch(bp + 32, 0, 3);    // WGP-scope prefetch, next K
        }
#pragma unroll
        for (int i = 0; i < 4; ++i)
#pragma unroll
            for (int j = 0; j < 4; ++j)
                acc[i * 4 + j] = wmma_bf16(a[i].v, b[j], acc[i * 4 + j]);
    }

#pragma unroll
    for (int i = 0; i < 4; ++i)
#pragma unroll
        for (int j = 0; j < 4; ++j) {
            f32x8 cacc = acc[i * 4 + j];
            int col = n0 + j * 16 + l16;
#pragma unroll
            for (int r = 0; r < 8; ++r) {
                int row = m0 + i * 16 + half * 8 + r;
                C[(size_t)row * N + col] = cacc[r];
            }
        }
}

// ---------------------------------------------------------------------------
// 4) RoPE on Q:  Qf f32 [B*S, NH*HD] -> Qbf bf16 [B, NH, S, HD]
// ---------------------------------------------------------------------------
__global__ void k_rope_q(const float* __restrict__ Qf,
                         __hip_bfloat16* __restrict__ Qbf) {
    unsigned idx = blockIdx.x * blockDim.x + threadIdx.x;   // < 2^23
    int d = idx & 255;
    int s = (idx >> 8) & 2047;
    int h = (idx >> 19) & 7;
    int b = idx >> 22;
    size_t base = (size_t)(b * Sc + s) * HIDc + h * HDc;
    float x = Qf[base + d];
    float partner = (d < 128) ? -Qf[base + d + 128] : Qf[base + d - 128];
    int dh = d & 127;
    float invf = __powf(10000.0f, -(float)(2 * dh) * (1.0f / 256.0f));
    float ang = (float)s * invf;
    float sn, cs;
    __sincosf(ang, &sn, &cs);
    Qbf[idx] = __float2bfloat16(x * cs + partner * sn);
}

// 5) RoPE on K:  Kf f32 [B*S, HD] -> Kbf bf16 [B, S, HD]
__global__ void k_rope_k(const float* __restrict__ Kf,
                         __hip_bfloat16* __restrict__ Kbf) {
    unsigned idx = blockIdx.x * blockDim.x + threadIdx.x;   // < 2^20
    int d = idx & 255;
    int s = (idx >> 8) & 2047;
    float x = Kf[idx];
    size_t base = idx - d;
    float partner = (d < 128) ? -Kf[base + d + 128] : Kf[base + d - 128];
    int dh = d & 127;
    float invf = __powf(10000.0f, -(float)(2 * dh) * (1.0f / 256.0f));
    float ang = (float)s * invf;
    float sn, cs;
    __sincosf(ang, &sn, &cs);
    Kbf[idx] = __float2bfloat16(x * cs + partner * sn);
}

// 6) V transpose: Vf f32 [B*S, HD] -> Vtbf bf16 [B, HD, S]
__global__ void k_vtrans(const float* __restrict__ Vf,
                         __hip_bfloat16* __restrict__ Vtbf) {
    unsigned idx = blockIdx.x * blockDim.x + threadIdx.x;   // < 2^20
    int d = idx & 255;
    int s = (idx >> 8) & 2047;
    int b = idx >> 19;
    Vtbf[((size_t)b * HDc + d) * Sc + s] = __float2bfloat16(Vf[idx]);
}

// ---------------------------------------------------------------------------
// 7) Flash attention: one wave per (b, h, 16-query tile). Online softmax.
//    Q: [B,NH,S,HD] bf16   K: [B,S,HD] bf16   Vt: [B,HD,S] bf16
//    Out: Abf [B*S, NH*HD] bf16 (GEMM-A layout for the O-projection)
// ---------------------------------------------------------------------------
__global__ __launch_bounds__(128) void k_attn(
        const __hip_bfloat16* __restrict__ Qbf,
        const __hip_bfloat16* __restrict__ Kbf,
        const __hip_bfloat16* __restrict__ Vtbf,
        __hip_bfloat16* __restrict__ Abf) {
    __shared__ __hip_bfloat16 Pbuf[4][16][32];

    int lane = threadIdx.x & 31;
    int wIn  = threadIdx.x >> 5;
    int wave = blockIdx.x * 4 + wIn;
    int qt = wave & 127;                  // S/16 = 128
    int h  = (wave >> 7) & 7;
    int b  = wave >> 10;
    int q0 = qt * 16;
    int half = lane >> 4, l16 = lane & 15;

    const __bf16* Q  = (const __bf16*)Qbf  + (size_t)(b * NHc + h) * Sc * HDc;
    const __bf16* Kk = (const __bf16*)Kbf  + (size_t)b * Sc * HDc;
    const __bf16* Vt = (const __bf16*)Vtbf + (size_t)b * HDc * Sc;

    // Preload 16x256 Q tile as 8 A-fragments (16x32 each)
    AFrag qa[8];
#pragma unroll
    for (int kc = 0; kc < 8; ++kc) {
        const __bf16* qp = Q + (size_t)(q0 + l16) * HDc + kc * 32 + half * 16;
        qa[kc].h[0] = *(const bf16x8*)qp;
        qa[kc].h[1] = *(const bf16x8*)(qp + 8);
    }

    f32x8 acc[16];
#pragma unroll
    for (int t = 0; t < 16; ++t) acc[t] = (f32x8){0,0,0,0,0,0,0,0};
    float mrow[8], lrow[8];
#pragma unroll
    for (int r = 0; r < 8; ++r) { mrow[r] = -1e30f; lrow[r] = 0.0f; }

    const float scale = 0.0625f;          // 1/sqrt(256)
    int kend = q0 + 16;                   // causal: keys <= last query row

    for (int k0 = 0; k0 < kend; k0 += 32) {
        // prefetch next key block (WGP scope) while this one computes
        if (k0 + 32 < kend) {
            __builtin_prefetch(Kk + (size_t)(k0 + 32 + l16) * HDc, 0, 3);
            __builtin_prefetch(Vt + (size_t)(l16 * 16) * Sc + k0 + 32, 0, 3);
        }
        // ---- scores: two 16x16 tiles over a 32-key block ------------------
        f32x8 s0 = (f32x8){0,0,0,0,0,0,0,0};
        f32x8 s1 = (f32x8){0,0,0,0,0,0,0,0};
#pragma unroll
        for (int kc = 0; kc < 8; ++kc) {
            bf16x16 b0 = *(const bf16x16*)(Kk + (size_t)(k0 + l16) * HDc +
                                           kc * 32 + half * 16);
            bf16x16 b1 = *(const bf16x16*)(Kk + (size_t)(k0 + 16 + l16) * HDc +
                                           kc * 32 + half * 16);
            s0 = wmma_bf16(qa[kc].v, b0, s0);
            s1 = wmma_bf16(qa[kc].v, b1, s1);
        }

        // ---- online softmax (per-row stats in each 16-lane half) ----------
        float corr[8];
#pragma unroll
        for (int r = 0; r < 8; ++r) {
            int row = q0 + half * 8 + r;
            float v0 = s0[r] * scale;
            float v1 = s1[r] * scale;
            if (k0 + l16 > row)      v0 = -1e30f;
            if (k0 + 16 + l16 > row) v1 = -1e30f;
            float mx = fmaxf(v0, v1);
#pragma unroll
            for (int dlt = 1; dlt < 16; dlt <<= 1)
                mx = fmaxf(mx, __shfl_xor(mx, dlt, 32));
            float mnew = fmaxf(mrow[r], mx);
            float c = __expf(mrow[r] - mnew);
            float p0 = __expf(v0 - mnew);
            float p1 = __expf(v1 - mnew);
            float ps = p0 + p1;
#pragma unroll
            for (int dlt = 1; dlt < 16; dlt <<= 1)
                ps += __shfl_xor(ps, dlt, 32);
            lrow[r] = lrow[r] * c + ps;
            mrow[r] = mnew;
            corr[r] = c;
            Pbuf[wIn][half * 8 + r][l16]      = __float2bfloat16(p0);
            Pbuf[wIn][half * 8 + r][16 + l16] = __float2bfloat16(p1);
        }
#pragma unroll
        for (int t = 0; t < 16; ++t)
#pragma unroll
            for (int r = 0; r < 8; ++r) acc[t][r] *= corr[r];

        // ---- P (C-layout) -> A-layout via per-wave LDS --------------------
        AFrag pa;
        const __hip_bfloat16* pl = &Pbuf[wIn][l16][half * 16];
        pa.h[0] = *(const bf16x8*)pl;
        pa.h[1] = *(const bf16x8*)(pl + 8);

        // ---- acc += P @ V  (16 output d-tiles of 16) ----------------------
#pragma unroll
        for (int t = 0; t < 16; ++t) {
            bf16x16 bv = *(const bf16x16*)(Vt + (size_t)(t * 16 + l16) * Sc +
                                           k0 + half * 16);
            acc[t] = wmma_bf16(pa.v, bv, acc[t]);
        }
    }

    // ---- normalize + store bf16 into O-projection A buffer ---------------
#pragma unroll
    for (int t = 0; t < 16; ++t) {
        int col = h * HDc + t * 16 + l16;
#pragma unroll
        for (int r = 0; r < 8; ++r) {
            size_t row = (size_t)b * Sc + q0 + half * 8 + r;
            Abf[row * (size_t)HIDc + col] =
                __float2bfloat16(acc[t][r] / lrow[r]);
        }
    }
}

// ---------------------------------------------------------------------------
// Host-side orchestration
// ---------------------------------------------------------------------------
extern "C" void kernel_launch(void* const* d_in, const int* in_sizes, int n_in,
                              void* d_out, int out_size, void* d_ws, size_t ws_size,
                              hipStream_t stream) {
    const float* hs = (const float*)d_in[0];
    // d_in[1] = attention_mask (causal, reproduced analytically) — unused
    const float* Wq = (const float*)d_in[2];
    const float* Wk = (const float*)d_in[3];
    const float* Wv = (const float*)d_in[4];
    const float* Wo = (const float*)d_in[5];
    float* out = (float*)d_out;

    uint8_t* w = (uint8_t*)d_ws;
    auto carve = [&](size_t bytes) -> void* {
        void* p = (void*)w;
        w += (bytes + 255) & ~(size_t)255;
        return p;
    };
    __hip_bfloat16* Xbf  = (__hip_bfloat16*)carve((size_t)Mrows * HIDc * 2);
    __hip_bfloat16* Wqt  = (__hip_bfloat16*)carve((size_t)HIDc * HIDc * 2);
    __hip_bfloat16* Wkt  = (__hip_bfloat16*)carve((size_t)HDc * HIDc * 2);
    __hip_bfloat16* Wvt  = (__hip_bfloat16*)carve((size_t)HDc * HIDc * 2);
    __hip_bfloat16* Wot  = (__hip_bfloat16*)carve((size_t)HIDc * HIDc * 2);
    float*          Qf   = (float*)carve((size_t)Mrows * HIDc * 4);
    float*          Kf   = (float*)carve((size_t)Mrows * HDc * 4);
    float*          Vf   = (float*)carve((size_t)Mrows * HDc * 4);
    __hip_bfloat16* Qbf  = (__hip_bfloat16*)carve((size_t)Bc * NHc * Sc * HDc * 2);
    __hip_bfloat16* Kbf  = (__hip_bfloat16*)carve((size_t)Bc * Sc * HDc * 2);
    __hip_bfloat16* Vtbf = (__hip_bfloat16*)carve((size_t)Bc * HDc * Sc * 2);
    __hip_bfloat16* Abf  = (__hip_bfloat16*)carve((size_t)Mrows * HIDc * 2);

    const int T = 256;
    // 1) activations -> bf16
    k_conv_bf16<<<(Mrows * HIDc) / T, T, 0, stream>>>(hs, Xbf, Mrows * HIDc);
    // 2) weights -> bf16, transposed to [N][K]
    k_wtrans<<<(HIDc * HIDc) / T, T, 0, stream>>>(Wq, Wqt, HIDc, HIDc);
    k_wtrans<<<(HIDc * HDc)  / T, T, 0, stream>>>(Wk, Wkt, HIDc, HDc);
    k_wtrans<<<(HIDc * HDc)  / T, T, 0, stream>>>(Wv, Wvt, HIDc, HDc);
    k_wtrans<<<(HIDc * HIDc) / T, T, 0, stream>>>(Wo, Wot, HIDc, HIDc);
    // 3) projections
    {
        // Q-proj: async-LDS staged GEMM, block = 64x512 tile
        int blksQ = (Mrows / 64) * (HIDc / 512);     // 256
        k_gemm_lds<<<blksQ, 256, 0, stream>>>(Xbf, Wqt, Qf, Mrows, HIDc, HIDc);
        // K/V-proj: direct GEMM (N=256)
        int wavesKV = (Mrows / 64) * (HDc / 64);     // 256
        k_gemm_bf16<<<wavesKV / 8, 256, 0, stream>>>(Xbf, Wkt, Kf, Mrows, HDc, HIDc);
        k_gemm_bf16<<<wavesKV / 8, 256, 0, stream>>>(Xbf, Wvt, Vf, Mrows, HDc, HIDc);
    }
    // 4) RoPE + layout conversions
    k_rope_q<<<(Bc * NHc * Sc * HDc) / T, T, 0, stream>>>(Qf, Qbf);
    k_rope_k<<<(Bc * Sc * HDc) / T, T, 0, stream>>>(Kf, Kbf);
    k_vtrans<<<(Bc * Sc * HDc) / T, T, 0, stream>>>(Vf, Vtbf);
    // 5) flash attention: 2048 waves, 4 waves/block
    k_attn<<<512, 128, 0, stream>>>(Qbf, Kbf, Vtbf, Abf);
    // 6) output projection (async-LDS staged) -> f32 result
    {
        int blksO = (Mrows / 64) * (HIDc / 512);     // 256
        k_gemm_lds<<<blksO, 256, 0, stream>>>(Abf, Wot, out, Mrows, HIDc, HIDc);
    }
    (void)in_sizes; (void)n_in; (void)out_size; (void)ws_size;
}
